// CATAttnBlock_90520730730919
// MI455X (gfx1250) — matrix operand order repaired and checked by reading, not compiled, not run on hardware
//
#include <hip/hip_runtime.h>
#include <hip/hip_bf16.h>

typedef __attribute__((ext_vector_type(16))) _Float16 v16h;
typedef __attribute__((ext_vector_type(8)))  float    v8f;

// ---------------------------------------------------------------------------
// WMMA helpers (gfx1250 wave32, v_wmma_f32_16x16x32_f16)
// ---------------------------------------------------------------------------
__device__ __forceinline__ v8f wmma_f16(v16h a, v16h b, v8f c) {
  // (neg_a, A, neg_b, B, c_mod, C, reuse_a, reuse_b)
  return __builtin_amdgcn_wmma_f32_16x16x32_f16(false, a, false, b, (short)0, c,
                                                false, false);
}

// A fragment: A stored row-major [M][K], 16x32 tile at (m0,k0).
// Layout: lane m = lane&15; VGPR0..7 hold K = kb+0..15,
// kb = (lane>>4)*16.
__device__ __forceinline__ v16h load_a(const _Float16* A, int ld, int m0, int k0) {
  const int lane = threadIdx.x & 31;
  const int m  = m0 + (lane & 15);
  const int kb = k0 + ((lane >> 4) << 4);
  const _Float16* p = A + (size_t)m * ld + kb;
  v16h a;
#pragma unroll
  for (int j = 0; j < 8; ++j) a[j] = p[j];
#pragma unroll
  for (int j = 0; j < 8; ++j) a[8 + j] = p[8 + j];
  return a;
}

// B fragment from transposed storage Bt[N][K] (i.e. column-major B), 32x16 tile.
// lane n = lane&15; K = kb + j, kb = (lane>>4)*16.
__device__ __forceinline__ v16h load_bt(const _Float16* Bt, int ld, int n0, int k0) {
  const int lane = threadIdx.x & 31;
  const int n  = n0 + (lane & 15);
  const int kb = k0 + ((lane >> 4) << 4);
  const _Float16* p = Bt + (size_t)n * ld + kb;
  v16h b;
#pragma unroll
  for (int j = 0; j < 16; ++j) b[j] = p[j];
  return b;
}

// B fragment from row-major storage B[K][N] (k-major), 32x16 tile at (k0,n0).
__device__ __forceinline__ v16h load_bk(const _Float16* B, int ld, int k0, int n0) {
  const int lane = threadIdx.x & 31;
  const int n  = n0 + (lane & 15);
  const int kb = k0 + ((lane >> 4) << 4);
  const _Float16* p = B + (size_t)kb * ld + n;
  v16h b;
#pragma unroll
  for (int j = 0; j < 16; ++j) b[j] = p[(size_t)j * ld];
  return b;
}

// C/D tile: lane holds column n = lane&15; VGPR r holds row m = m0 + (lane>>4)*8 + r.
__device__ __forceinline__ void store_tile_f16(_Float16* D, int ld, int m0, int n0, v8f c) {
  const int lane = threadIdx.x & 31;
  const int n = n0 + (lane & 15);
  const int m = m0 + ((lane >> 4) << 3);
#pragma unroll
  for (int r = 0; r < 8; ++r) D[(size_t)(m + r) * ld + n] = (_Float16)c[r];
}

__device__ __forceinline__ void store_tile_f16_bias(_Float16* D, int ld, int m0, int n0,
                                                    v8f c, float bv) {
  const int lane = threadIdx.x & 31;
  const int n = n0 + (lane & 15);
  const int m = m0 + ((lane >> 4) << 3);
#pragma unroll
  for (int r = 0; r < 8; ++r) D[(size_t)(m + r) * ld + n] = (_Float16)(c[r] + bv);
}

// ---------------------------------------------------------------------------
// Problem constants
// ---------------------------------------------------------------------------
#define BB   4
#define CC   256
#define HH   112
#define WW2  112
#define PSZ  7
#define NTOK 49
#define ND   64     // padded tokens
#define NHD  8
#define DHD  32
#define NPW  256    // windows per image (16x16)
#define BN   1024   // BB*NPW
#define BC   1024   // BB*CC
#define OUTN 12845056  // BB*CC*HH*WW2

// ---------------------------------------------------------------------------
// Prep: transpose f32 weights to f16 [out][in]
// ---------------------------------------------------------------------------
__global__ void k_transpose_to_f16(const float* __restrict__ src,
                                   _Float16* __restrict__ dst,
                                   int rows_in, int cols_out) {
  int i = blockIdx.x * blockDim.x + threadIdx.x;
  int total = rows_in * cols_out;
  if (i >= total) return;
  int o = i / rows_in;
  int k = i - o * rows_in;
  dst[i] = (_Float16)src[(size_t)k * cols_out + o];
}

// Prep: cross-attn weights split (q/k/v), padded 49->64, transposed [out][in]; padded biases.
__global__ void k_prep_cross(const float* __restrict__ qw, const float* __restrict__ qb,
                             const float* __restrict__ pw, const float* __restrict__ pb,
                             _Float16* __restrict__ Wq, _Float16* __restrict__ Wk,
                             _Float16* __restrict__ Wv, _Float16* __restrict__ Wp,
                             float* __restrict__ bq, float* __restrict__ bp) {
  int i = blockIdx.x * blockDim.x + threadIdx.x;
  if (i >= 4096) return;
  int o = i >> 6, k = i & 63;
  bool valid = (o < NTOK) && (k < NTOK);
  Wq[i] = valid ? (_Float16)qw[k * 147 + o]       : (_Float16)0.f;
  Wk[i] = valid ? (_Float16)qw[k * 147 + 49 + o]  : (_Float16)0.f;
  Wv[i] = valid ? (_Float16)qw[k * 147 + 98 + o]  : (_Float16)0.f;
  Wp[i] = valid ? (_Float16)pw[k * 49 + o]        : (_Float16)0.f;
  if (i < 192) { int mat = i >> 6, j = i & 63; bq[i] = (j < NTOK) ? qb[mat * 49 + j] : 0.f; }
  if (i < 64)  { bp[i] = (i < NTOK) ? pb[i] : 0.f; }
}

// ---------------------------------------------------------------------------
// Stage 1: LayerNorm over C + window partition -> fp16 patches
//   P16  : (BN, 64, 256)    token-major (padded token rows zeroed)
//   X16c : (B*C, 256, 64)   channel-major for cross attention (padded cols zeroed)
// One wave per token; block = one window.
// ---------------------------------------------------------------------------
__global__ void __launch_bounds__(256) k_ln_partition(
    const float* __restrict__ x, const float* __restrict__ gamma,
    const float* __restrict__ beta, _Float16* __restrict__ P16,
    _Float16* __restrict__ X16c) {
  const int wnd = blockIdx.x;
  const int b = wnd >> 8, p = wnd & 255;
  const int py = p >> 4, px = p & 15;
  const int lane = threadIdx.x & 31, wave = threadIdx.x >> 5;
  for (int iter = 0; iter < 8; ++iter) {
    const int t = iter * 8 + wave;  // 0..63
    _Float16* prow = P16 + (size_t)wnd * ND * CC + (size_t)t * CC;
    if (t < NTOK) {
      const int ty = t / 7, tx = t - ty * 7;
      const int h = py * 7 + ty, w = px * 7 + tx;
      const float* xp = x + ((size_t)b * CC * HH + h) * WW2 + w;
      float vals[8];
      float s = 0.f, s2 = 0.f;
#pragma unroll
      for (int j = 0; j < 8; ++j) {
        int c = lane + 32 * j;
        float v = xp[(size_t)c * (HH * WW2)];
        vals[j] = v; s += v; s2 += v * v;
      }
#pragma unroll
      for (int off = 1; off < 32; off <<= 1) {
        s  += __shfl_xor(s, off, 32);
        s2 += __shfl_xor(s2, off, 32);
      }
      const float mu  = s * (1.f / 256.f);
      const float var = s2 * (1.f / 256.f) - mu * mu;
      const float rinv = rsqrtf(var + 1e-5f);
#pragma unroll
      for (int j = 0; j < 8; ++j) {
        int c = lane + 32 * j;
        float y = (vals[j] - mu) * rinv * gamma[c] + beta[c];
        prow[c] = (_Float16)y;
        X16c[(size_t)(b * CC + c) * (NPW * ND) + (size_t)p * ND + t] = (_Float16)y;
      }
    } else {  // zero padding rows
#pragma unroll
      for (int j = 0; j < 8; ++j) {
        int c = lane + 32 * j;
        prow[c] = (_Float16)0.f;
        X16c[(size_t)(b * CC + c) * (NPW * ND) + (size_t)p * ND + t] = (_Float16)0.f;
      }
    }
  }
}

// ---------------------------------------------------------------------------
// Stage 2: local windowed attention (8 heads, d=32, Swin relative-position bias)
// One block per window; wave w = head w in the attention phase.
// LDS: Xs 64x264 | QKV 64x776 | P 8x(64x72) | O 64x264 | rpb 169x8 f32
// Output (fp16, padded rows zeroed) aliases P16.
// ---------------------------------------------------------------------------
#define LA_XS_LD 264
#define LA_QK_LD 776
#define LA_PS_LD 72
#define LA_OS_LD 264

__global__ void __launch_bounds__(256) k_local_attn(
    const _Float16* P16, const _Float16* __restrict__ Wqkv_t,
    const float* __restrict__ qkv_b, const _Float16* __restrict__ Wpl_t,
    const float* __restrict__ proj_b, const float* __restrict__ rpb,
    _Float16* Lout) {
  extern __shared__ _Float16 sm[];
  _Float16* Xs = sm;                           // 64 x 264
  _Float16* QK = Xs + 64 * LA_XS_LD;           // 64 x 776
  _Float16* Ps = QK + 64 * LA_QK_LD;           // 8 x 64 x 72
  _Float16* Os = Ps + 8 * 64 * LA_PS_LD;       // 64 x 264
  float* rpbs  = (float*)(Os + 64 * LA_OS_LD); // 169*8

  const int wnd = blockIdx.x;
  const int tid = threadIdx.x, lane = tid & 31, wave = tid >> 5;

  for (int i = tid; i < 169 * NHD; i += 256) rpbs[i] = rpb[i];
  const _Float16* src = P16 + (size_t)wnd * ND * CC;
  for (int i = tid; i < ND * CC; i += 256) {
    int r = i >> 8, c = i & 255;
    Xs[r * LA_XS_LD + c] = src[i];
  }
  __syncthreads();

  // ---- QKV GEMM: (64x256) @ (256x768), wave owns 96 output columns ----
  for (int ct = 0; ct < 6; ++ct) {
    const int n0 = wave * 96 + ct * 16;
    const float bv = qkv_b[n0 + (lane & 15)];
    for (int rt = 0; rt < 4; ++rt) {
      v8f acc = {};
#pragma unroll
      for (int kc = 0; kc < 8; ++kc)
        acc = wmma_f16(load_a(Xs, LA_XS_LD, rt * 16, kc * 32),
                       load_bt(Wqkv_t, 256, n0, kc * 32), acc);
      store_tile_f16_bias(QK, LA_QK_LD, rt * 16, n0, acc, bv);
    }
  }
  __syncthreads();

  // ---- attention: wave = head ----
  {
    const int h = wave;
    const int qc = h * DHD, kcol = CC + h * DHD, vcol = 2 * CC + h * DHD;
    const float scale = 0.17677669529663687f;  // 1/sqrt(32)
    v8f s[4][4];
#pragma unroll
    for (int rt = 0; rt < 4; ++rt)
#pragma unroll
      for (int ct = 0; ct < 4; ++ct) {
        v8f acc = {};
        acc = wmma_f16(load_a(QK, LA_QK_LD, rt * 16, qc),
                       load_bt(QK, LA_QK_LD, ct * 16, kcol), acc);
        s[rt][ct] = acc;
      }
    // scale + relative-position bias + key-padding mask
    const int nlo = lane & 15;
    const int mbse = (lane >> 4) << 3;
#pragma unroll
    for (int rt = 0; rt < 4; ++rt)
#pragma unroll
      for (int ct = 0; ct < 4; ++ct) {
        const int n = ct * 16 + nlo;
#pragma unroll
        for (int r = 0; r < 8; ++r) {
          const int m = rt * 16 + mbse + r;
          float v = s[rt][ct][r] * scale;
          if (n >= NTOK) v = -1e30f;
          else if (m < NTOK) {
            const int qy = m / 7, qx = m - qy * 7;
            const int ky = n / 7, kx = n - ky * 7;
            const int idx = (qy - ky + 6) * 13 + (qx - kx + 6);
            v += rpbs[idx * NHD + h];
          }
          s[rt][ct][r] = v;
        }
      }
    // softmax over keys (cols): cross-tile + 16-lane reduction
#pragma unroll
    for (int rt = 0; rt < 4; ++rt) {
#pragma unroll
      for (int r = 0; r < 8; ++r) {
        float mx = -3.4e38f;
#pragma unroll
        for (int ct = 0; ct < 4; ++ct) mx = fmaxf(mx, s[rt][ct][r]);
#pragma unroll
        for (int off = 1; off < 16; off <<= 1) mx = fmaxf(mx, __shfl_xor(mx, off, 32));
        float sum = 0.f;
#pragma unroll
        for (int ct = 0; ct < 4; ++ct) {
          float e = __expf(s[rt][ct][r] - mx);
          s[rt][ct][r] = e; sum += e;
        }
#pragma unroll
        for (int off = 1; off < 16; off <<= 1) sum += __shfl_xor(sum, off, 32);
        const float inv = 1.f / sum;
#pragma unroll
        for (int ct = 0; ct < 4; ++ct) s[rt][ct][r] *= inv;
      }
    }
    _Float16* Pw = Ps + (size_t)wave * 64 * LA_PS_LD;
#pragma unroll
    for (int rt = 0; rt < 4; ++rt)
#pragma unroll
      for (int ct = 0; ct < 4; ++ct)
        store_tile_f16(Pw, LA_PS_LD, rt * 16, ct * 16, s[rt][ct]);
    // O = P @ V  (64x64)@(64x32)
#pragma unroll
    for (int rt = 0; rt < 4; ++rt)
#pragma unroll
      for (int cd = 0; cd < 2; ++cd) {
        v8f o = {};
#pragma unroll
        for (int kk = 0; kk < 2; ++kk)
          o = wmma_f16(load_a(Pw, LA_PS_LD, rt * 16, kk * 32),
                       load_bk(QK, LA_QK_LD, kk * 32, vcol + cd * 16), o);
        store_tile_f16(Os, LA_OS_LD, rt * 16, h * DHD + cd * 16, o);
      }
  }
  __syncthreads();

  // ---- output proj: (64x256)@(256x256), wave owns 32 cols; pad rows -> 0 ----
  _Float16* dst = Lout + (size_t)wnd * ND * CC;
  for (int cc = 0; cc < 2; ++cc) {
    const int n0 = wave * 32 + cc * 16;
    const int n = n0 + (lane & 15);
    const float bv = proj_b[n];
    for (int rt = 0; rt < 4; ++rt) {
      v8f acc = {};
#pragma unroll
      for (int kc = 0; kc < 8; ++kc)
        acc = wmma_f16(load_a(Os, LA_OS_LD, rt * 16, kc * 32),
                       load_bt(Wpl_t, 256, n0, kc * 32), acc);
      const int mb = rt * 16 + ((lane >> 4) << 3);
#pragma unroll
      for (int r = 0; r < 8; ++r) {
        const int m = mb + r;
        dst[(size_t)m * CC + n] = (m < NTOK) ? (_Float16)(acc[r] + bv) : (_Float16)0.f;
      }
    }
  }
}

// ---------------------------------------------------------------------------
// Stage 3: cross-channel attention. batch = B*C, seq = 256 windows, dim = 49->64.
// One block per (b,c); wave owns sequence row-tiles {w, w+8}.
// LDS: X 256x72 | Q/K/V 256x72 | P 8x(16x264) | O 256x72
// Output (fp16) aliases X16c.
// ---------------------------------------------------------------------------
#define CA_X_LD 72
#define CA_PS_LD 264

__global__ void __launch_bounds__(256) k_cross_attn(
    const _Float16* X16c,
    const _Float16* __restrict__ Wqc, const _Float16* __restrict__ Wkc,
    const _Float16* __restrict__ Wvc, const _Float16* __restrict__ Wpc,
    const float* __restrict__ bqkv, const float* __restrict__ bpc,
    _Float16* Cout) {
  extern __shared__ _Float16 sm[];
  _Float16* Xs = sm;                          // 256 x 72
  _Float16* Qs = Xs + 256 * CA_X_LD;
  _Float16* Ks = Qs + 256 * CA_X_LD;
  _Float16* Vs = Ks + 256 * CA_X_LD;
  _Float16* Ps = Vs + 256 * CA_X_LD;          // 8 x 16 x 264
  _Float16* Os = Ps + 8 * 16 * CA_PS_LD;      // 256 x 72

  const int bc = blockIdx.x;
  const int tid = threadIdx.x, lane = tid & 31, wave = tid >> 5;

  const _Float16* src = X16c + (size_t)bc * NPW * ND;
  for (int i = tid; i < NPW * ND; i += 256) {
    int r = i >> 6, c = i & 63;
    Xs[r * CA_X_LD + c] = src[i];
  }
  __syncthreads();

  // ---- QKV: three (256x64)@(64x64) GEMMs ----
  for (int mat = 0; mat < 3; ++mat) {
    const _Float16* W = (mat == 0) ? Wqc : (mat == 1) ? Wkc : Wvc;
    _Float16* D = (mat == 0) ? Qs : (mat == 1) ? Ks : Vs;
    for (int rr = 0; rr < 2; ++rr) {
      const int rt = wave + rr * 8;
      for (int ct = 0; ct < 4; ++ct) {
        v8f acc = {};
#pragma unroll
        for (int kc = 0; kc < 2; ++kc)
          acc = wmma_f16(load_a(Xs, CA_X_LD, rt * 16, kc * 32),
                         load_bt(W, 64, ct * 16, kc * 32), acc);
        const float bv = bqkv[mat * 64 + ct * 16 + (lane & 15)];
        store_tile_f16_bias(D, CA_X_LD, rt * 16, ct * 16, acc, bv);
      }
    }
  }
  __syncthreads();

  // ---- S = QK^T / 7, softmax over 256 keys, O = P V ----
  {
    const float scale = 1.f / 7.f;  // 1/sqrt(49)
    _Float16* Pw = Ps + (size_t)wave * 16 * CA_PS_LD;
    for (int pass = 0; pass < 2; ++pass) {
      const int rt = pass * 8 + wave;
      v8f s[16];
#pragma unroll
      for (int ct = 0; ct < 16; ++ct) {
        v8f acc = {};
#pragma unroll
        for (int kc = 0; kc < 2; ++kc)
          acc = wmma_f16(load_a(Qs, CA_X_LD, rt * 16, kc * 32),
                         load_bt(Ks, CA_X_LD, ct * 16, kc * 32), acc);
        s[ct] = acc;
      }
#pragma unroll
      for (int ct = 0; ct < 16; ++ct)
#pragma unroll
        for (int r = 0; r < 8; ++r) s[ct][r] *= scale;
#pragma unroll
      for (int r = 0; r < 8; ++r) {
        float mx = -3.4e38f;
#pragma unroll
        for (int ct = 0; ct < 16; ++ct) mx = fmaxf(mx, s[ct][r]);
#pragma unroll
        for (int off = 1; off < 16; off <<= 1) mx = fmaxf(mx, __shfl_xor(mx, off, 32));
        float sum = 0.f;
#pragma unroll
        for (int ct = 0; ct < 16; ++ct) {
          float e = __expf(s[ct][r] - mx);
          s[ct][r] = e; sum += e;
        }
#pragma unroll
        for (int off = 1; off < 16; off <<= 1) sum += __shfl_xor(sum, off, 32);
        const float inv = 1.f / sum;
#pragma unroll
        for (int ct = 0; ct < 16; ++ct) s[ct][r] *= inv;
      }
#pragma unroll
      for (int ct = 0; ct < 16; ++ct)
        store_tile_f16(Pw, CA_PS_LD, 0, ct * 16, s[ct]);
      for (int cd = 0; cd < 4; ++cd) {
        v8f o = {};
#pragma unroll
        for (int kc = 0; kc < 8; ++kc)
          o = wmma_f16(load_a(Pw, CA_PS_LD, 0, kc * 32),
                       load_bk(Vs, CA_X_LD, kc * 32, cd * 16), o);
        store_tile_f16(Os, CA_X_LD, rt * 16, cd * 16, o);
      }
    }
  }
  __syncthreads();

  // ---- proj: (256x64)@(64x64) -> Cout (padded cols are exactly zero) ----
  _Float16* dst = Cout + (size_t)bc * NPW * ND;
  for (int rr = 0; rr < 2; ++rr) {
    const int rt = wave + rr * 8;
    for (int ct = 0; ct < 4; ++ct) {
      v8f acc = {};
#pragma unroll
      for (int kc = 0; kc < 2; ++kc)
        acc = wmma_f16(load_a(Os, CA_X_LD, rt * 16, kc * 32),
                       load_bt(Wpc, 64, ct * 16, kc * 32), acc);
      const int n = ct * 16 + (lane & 15);
      const float bv = bpc[n];
      const int mb = rt * 16 + ((lane >> 4) << 3);
#pragma unroll
      for (int r = 0; r < 8; ++r)
        dst[(size_t)(mb + r) * ND + n] = (_Float16)(acc[r] + bv);
    }
  }
}

// ---------------------------------------------------------------------------
// Stage 4: concat(local, cross) @ proj_w (512x256) and scatter to (B,C,H,W).
// One block per window; A staged 64x512 f16 in LDS.
// ---------------------------------------------------------------------------
#define FA_LD 520

__global__ void __launch_bounds__(256) k_final_proj(
    const _Float16* __restrict__ Lout, const _Float16* __restrict__ Cout,
    const _Float16* __restrict__ Wf_t, float* __restrict__ out) {
  extern __shared__ _Float16 sm[];
  _Float16* As = sm;  // 64 x 520
  const int wnd = blockIdx.x;
  const int b = wnd >> 8, p = wnd & 255;
  const int tid = threadIdx.x, lane = tid & 31, wave = tid >> 5;

  const _Float16* lsrc = Lout + (size_t)wnd * ND * CC;
  for (int i = tid; i < ND * CC; i += 256) {
    int r = i >> 8, c = i & 255;
    As[r * FA_LD + c] = lsrc[i];
  }
  {
    const int c = tid;  // 0..255
    const _Float16* csrc = Cout + ((size_t)(b * CC + c) * NPW + p) * ND;
#pragma unroll 4
    for (int t = 0; t < ND; ++t) As[t * FA_LD + CC + c] = csrc[t];
  }
  __syncthreads();

  const int py = p >> 4, px = p & 15;
  for (int cc = 0; cc < 2; ++cc) {
    const int n0 = wave * 32 + cc * 16;
    const int n = n0 + (lane & 15);
    for (int rt = 0; rt < 4; ++rt) {
      v8f acc = {};
#pragma unroll
      for (int kc = 0; kc < 16; ++kc)
        acc = wmma_f16(load_a(As, FA_LD, rt * 16, kc * 32),
                       load_bt(Wf_t, 512, n0, kc * 32), acc);
      const int mb = rt * 16 + ((lane >> 4) << 3);
#pragma unroll
      for (int r = 0; r < 8; ++r) {
        const int m = mb + r;
        if (m < NTOK) {
          const int ty = m / 7, tx = m - ty * 7;
          const int hh = py * 7 + ty, ww = px * 7 + tx;
          out[((size_t)(b * CC + n) * HH + hh) * WW2 + ww] = acc[r];
        }
      }
    }
  }
}

// Shortcut copy (second output of the tuple) — vectorized stream.
__global__ void k_copy4(const float4* __restrict__ src, float4* __restrict__ dst, int n4) {
  int i = blockIdx.x * blockDim.x + threadIdx.x;
  for (; i < n4; i += gridDim.x * blockDim.x) dst[i] = src[i];
}

// ---------------------------------------------------------------------------
// Launch
// ---------------------------------------------------------------------------
extern "C" void kernel_launch(void* const* d_in, const int* in_sizes, int n_in,
                              void* d_out, int out_size, void* d_ws, size_t ws_size,
                              hipStream_t stream) {
  const float* x        = (const float*)d_in[0];
  const float* gamma    = (const float*)d_in[1];
  const float* beta     = (const float*)d_in[2];
  const float* qkv_l_w  = (const float*)d_in[3];
  const float* qkv_l_b  = (const float*)d_in[4];
  const float* proj_l_w = (const float*)d_in[5];
  const float* proj_l_b = (const float*)d_in[6];
  const float* rpb      = (const float*)d_in[7];
  const float* qkv_c_w  = (const float*)d_in[8];
  const float* qkv_c_b  = (const float*)d_in[9];
  const float* proj_c_w = (const float*)d_in[10];
  const float* proj_c_b = (const float*)d_in[11];
  const float* proj_w   = (const float*)d_in[12];
  float* out = (float*)d_out;

  // Workspace layout (bytes). P16 is reused as the local-attn output and
  // X16c as the cross-attn output (block-local read-then-write, safe).
  char* ws = (char*)d_ws;
  _Float16* P16   = (_Float16*)(ws);                  // 1024*64*256 f16 = 32 MiB
  _Float16* X16c  = (_Float16*)(ws + 33554432ull);    // 1024*256*64 f16 = 32 MiB
  _Float16* Wqkv  = (_Float16*)(ws + 67108864ull);    // 768x256 f16
  _Float16* Wpl   = (_Float16*)(ws + 67502080ull);    // 256x256 f16
  _Float16* Wf    = (_Float16*)(ws + 67633152ull);    // 256x512 f16
  _Float16* Wqc   = (_Float16*)(ws + 67895296ull);    // 64x64 f16
  _Float16* Wkc   = Wqc + 4096;
  _Float16* Wvc   = Wkc + 4096;
  _Float16* Wpc   = Wvc + 4096;
  float*    bqkvc = (float*)(ws + 67928064ull);       // 192 f32
  float*    bpcp  = bqkvc + 192;                      // 64 f32

  constexpr size_t SM_LOCAL =
      (size_t)(64 * LA_XS_LD + 64 * LA_QK_LD + 8 * 64 * LA_PS_LD + 64 * LA_OS_LD) * 2 +
      169 * 8 * 4;                                         // 246048 B
  constexpr size_t SM_CROSS =
      (size_t)(4 * 256 * CA_X_LD + 8 * 16 * CA_PS_LD + 256 * CA_X_LD) * 2;  // 251904 B
  constexpr size_t SM_FINAL = (size_t)(64 * FA_LD) * 2;    // 66560 B

  (void)hipFuncSetAttribute((const void*)k_local_attn,
                            hipFuncAttributeMaxDynamicSharedMemorySize, (int)SM_LOCAL);
  (void)hipFuncSetAttribute((const void*)k_cross_attn,
                            hipFuncAttributeMaxDynamicSharedMemorySize, (int)SM_CROSS);
  (void)hipFuncSetAttribute((const void*)k_final_proj,
                            hipFuncAttributeMaxDynamicSharedMemorySize, (int)SM_FINAL);

  // Weight prep (L2-resident fp16, transposed [out][in])
  k_transpose_to_f16<<<(768 * 256 + 255) / 256, 256, 0, stream>>>(qkv_l_w, Wqkv, 256, 768);
  k_transpose_to_f16<<<(256 * 256 + 255) / 256, 256, 0, stream>>>(proj_l_w, Wpl, 256, 256);
  k_transpose_to_f16<<<(256 * 512 + 255) / 256, 256, 0, stream>>>(proj_w, Wf, 512, 256);
  k_prep_cross<<<16, 256, 0, stream>>>(qkv_c_w, qkv_c_b, proj_c_w, proj_c_b,
                                       Wqc, Wkc, Wvc, Wpc, bqkvc, bpcp);

  // Pipeline
  k_ln_partition<<<BN, 256, 0, stream>>>(x, gamma, beta, P16, X16c);
  k_local_attn<<<BN, 256, SM_LOCAL, stream>>>(P16, Wqkv, qkv_l_b, Wpl, proj_l_b, rpb,
                                              P16 /* Lout aliases P16 */);
  k_cross_attn<<<BC, 256, SM_CROSS, stream>>>(X16c, Wqc, Wkc, Wvc, Wpc, bqkvc, bpcp,
                                              X16c /* Cout aliases X16c */);
  k_final_proj<<<BN, 256, SM_FINAL, stream>>>(P16, X16c, Wf, out);

  // shortcut = x (second tuple output)
  k_copy4<<<2048, 256, 0, stream>>>((const float4*)x, (float4*)out + OUTN / 4, OUTN / 4);
}
